// layer_81398220194654
// MI455X (gfx1250) — matrix operand order repaired and checked by reading, not compiled, not run on hardware
//
#include <hip/hip_runtime.h>

// ---------------- constants ----------------
#define Bb   2
#define Ss   2048
#define Dd   1024
#define NHh  16
#define DHh  64
#define Ee   8
#define KTOP 2
#define HIDh 2048
#define Tt   (Bb * Ss)     // 4096 tokens

#define BM 128
#define BN 64
#define BK 32

typedef __bf16 bf16_t;
typedef __attribute__((ext_vector_type(16))) __bf16 v16bf;
typedef __attribute__((ext_vector_type(8)))  __bf16 v8bf;
typedef __attribute__((ext_vector_type(8)))  float  v8f;

__device__ __forceinline__ v8f zero_v8f() {
  v8f z;
#pragma unroll
  for (int i = 0; i < 8; ++i) z[i] = 0.f;
  return z;
}

// Build a 16-half fragment from two contiguous 8-half LDS runs.
__device__ __forceinline__ v16bf mk_frag(const bf16_t* p0, const bf16_t* p1) {
  v8bf lo = *(const v8bf*)p0;
  v8bf hi = *(const v8bf*)p1;
  v16bf r;
#pragma unroll
  for (int i = 0; i < 8; ++i) { r[i] = lo[i]; r[i + 8] = hi[i]; }
  return r;
}

__device__ __forceinline__ v8f wmma_bf16(v16bf a, v16bf b, v8f c) {
  return __builtin_amdgcn_wmma_f32_16x16x32_bf16(false, a, false, b, (short)0, c,
                                                 false, false);
}

// ---------------- RMSNorm (f32 in -> bf16 out) ----------------
__global__ __launch_bounds__(256) void rmsnorm_kernel(const float* __restrict__ x,
                                                      const float* __restrict__ w,
                                                      bf16_t* __restrict__ out,
                                                      float eps) {
  int row = blockIdx.x;
  int tid = threadIdx.x;
  const float* xr = x + (size_t)row * Dd;
  float4 xv = ((const float4*)xr)[tid];                // 4 elems/thread, D=1024
  float ss = xv.x * xv.x + xv.y * xv.y + xv.z * xv.z + xv.w * xv.w;
#pragma unroll
  for (int off = 16; off >= 1; off >>= 1) ss += __shfl_xor(ss, off, 32);
  __shared__ float red[8];
  __shared__ float rtot;
  if ((tid & 31) == 0) red[tid >> 5] = ss;
  __syncthreads();
  if (tid == 0) {
    float t = 0.f;
#pragma unroll
    for (int i = 0; i < 8; ++i) t += red[i];
    rtot = rsqrtf(t / (float)Dd + eps);
  }
  __syncthreads();
  float rs = rtot;
  bf16_t* orow = out + (size_t)row * Dd;
  const float* wr = w + tid * 4;
  orow[tid * 4 + 0] = (bf16_t)(xv.x * rs * wr[0]);
  orow[tid * 4 + 1] = (bf16_t)(xv.y * rs * wr[1]);
  orow[tid * 4 + 2] = (bf16_t)(xv.z * rs * wr[2]);
  orow[tid * 4 + 3] = (bf16_t)(xv.w * rs * wr[3]);
}

// ---------------- RoPE (in-place on bf16 Q or K) ----------------
__global__ __launch_bounds__(256) void rope_kernel(bf16_t* __restrict__ q,
                                                   const float* __restrict__ fc,
                                                   const float* __restrict__ fs) {
  int i = blockIdx.x * 256 + threadIdx.x;  // pair index, T*D/2 total
  int t   = i / (Dd / 2);
  int rem = i % (Dd / 2);
  int h = rem / (DHh / 2);
  int p = rem % (DHh / 2);
  int s = t % Ss;
  float c  = fc[s * (DHh / 2) + p];
  float sn = fs[s * (DHh / 2) + p];
  size_t base = (size_t)t * Dd + h * DHh + 2 * p;
  float qr = (float)q[base];
  float qi = (float)q[base + 1];
  q[base]     = (bf16_t)(qr * c - qi * sn);
  q[base + 1] = (bf16_t)(qr * sn + qi * c);
}

// ---------------- WMMA GEMM: C = A(bf16) x B(f32->bf16), tiled in LDS --------
// Software-pipelined: next tile's global loads issue before the WMMAs of the
// current tile, so HBM latency overlaps matrix math.
// mode 1: Cb[row]   = bf16(silu(acc))              (w1)
// mode 2: Cf[c_idx[row]] += c_scale[row]*acc       (w2, gathered accumulate)
// mode 3: Cf[row]   = add_src[row] + acc           (o-proj + residual)
// mode 4: Cb[row]   = bf16(acc)                    (QKV)
__global__ __launch_bounds__(256) void gemm_bf16(
    const bf16_t* __restrict__ A, int lda,
    const float* __restrict__ Bw, int ldb,
    float* __restrict__ Cf, bf16_t* __restrict__ Cb, int ldc,
    int Mconst, const int* __restrict__ Mptr, int N, int Kdim,
    const int* __restrict__ a_idx, const int* __restrict__ c_idx,
    const float* __restrict__ c_scale, const float* __restrict__ add_src,
    int mode) {
  int Mv = Mptr ? *Mptr : Mconst;
  int m0 = blockIdx.y * BM;
  if (m0 >= Mv) return;
  int n0 = blockIdx.x * BN;

  __shared__ bf16_t As[BM * BK];       // row-major, k-contig
  __shared__ bf16_t Bst[BN * BK];      // transposed: [n][k], k-contig

  int tid  = threadIdx.x;
  int lane = tid & 31;
  int w    = tid >> 5;                 // 8 waves
  int wm   = w & 3, wn = w >> 2;       // 4 x 2 wave grid -> 32x32 per wave
  int g    = lane >> 4, ln = lane & 15;

  // ---- hoisted, loop-invariant A-row gather (one index load per chunk) ----
  const bf16_t* aSrc[2];
  int aC8[2], aRow[2], bKr[2], bC4[2];
#pragma unroll
  for (int c = 0; c < 2; ++c) {
    int idx = tid + 256 * c;                 // 0..511
    int r = idx >> 2;                        // r < 128
    aC8[c] = idx & 3;
    aRow[c] = r;
    int gm = m0 + r;
    int src = (gm < Mv) ? (a_idx ? a_idx[gm] : gm) : 0;
    aSrc[c] = A + (size_t)src * lda + aC8[c] * 8;
    bKr[c] = idx >> 4;                       // < 32
    bC4[c] = idx & 15;                       // 16 float4 per row
  }

  v8bf   aReg[2];
  float4 bReg[2];
#pragma unroll
  for (int c = 0; c < 2; ++c) {              // prologue: tile k0 = 0
    aReg[c] = *(const v8bf*)(aSrc[c]);
    bReg[c] = *(const float4*)(Bw + (size_t)bKr[c] * ldb + n0 + bC4[c] * 4);
  }

  v8f acc[2][2];
#pragma unroll
  for (int mt = 0; mt < 2; ++mt)
#pragma unroll
    for (int nt = 0; nt < 2; ++nt) acc[mt][nt] = zero_v8f();

  for (int k0 = 0; k0 < Kdim; k0 += BK) {
    // ---- commit staged registers to LDS ----
#pragma unroll
    for (int c = 0; c < 2; ++c) {
      *(v8bf*)(As + aRow[c] * BK + aC8[c] * 8) = aReg[c];
      Bst[(bC4[c] * 4 + 0) * BK + bKr[c]] = (bf16_t)bReg[c].x;
      Bst[(bC4[c] * 4 + 1) * BK + bKr[c]] = (bf16_t)bReg[c].y;
      Bst[(bC4[c] * 4 + 2) * BK + bKr[c]] = (bf16_t)bReg[c].z;
      Bst[(bC4[c] * 4 + 3) * BK + bKr[c]] = (bf16_t)bReg[c].w;
    }
    __syncthreads();

    // ---- issue next tile's global loads (overlap with WMMA below) ----
    int kn = k0 + BK;
    if (kn < Kdim) {
#pragma unroll
      for (int c = 0; c < 2; ++c) {
        aReg[c] = *(const v8bf*)(aSrc[c] + kn);
        bReg[c] = *(const float4*)(Bw + (size_t)(kn + bKr[c]) * ldb + n0 + bC4[c] * 4);
      }
    }

    // ---- fragments + 4 WMMAs per wave ----
    v16bf af[2], bfr[2];
#pragma unroll
    for (int mt = 0; mt < 2; ++mt) {
      const bf16_t* ar = As + (wm * 32 + mt * 16 + ln) * BK;
      af[mt] = mk_frag(ar + 16 * g, ar + 16 * g + 8);
    }
#pragma unroll
    for (int nt = 0; nt < 2; ++nt) {
      const bf16_t* br = Bst + (wn * 32 + nt * 16 + ln) * BK + 16 * g;
      bfr[nt] = mk_frag(br, br + 8);
    }
#pragma unroll
    for (int mt = 0; mt < 2; ++mt)
#pragma unroll
      for (int nt = 0; nt < 2; ++nt)
        acc[mt][nt] = wmma_bf16(af[mt], bfr[nt], acc[mt][nt]);
    __syncthreads();
  }

  // ---- epilogue ----
#pragma unroll
  for (int mt = 0; mt < 2; ++mt)
#pragma unroll
    for (int nt = 0; nt < 2; ++nt)
#pragma unroll
      for (int vv = 0; vv < 8; ++vv) {
        int rl = wm * 32 + mt * 16 + vv + 8 * g;   // C layout: row = v + 8*(l/16)
        int gm = m0 + rl;
        if (gm >= Mv) continue;
        int col = n0 + wn * 32 + nt * 16 + ln;
        float val = acc[mt][nt][vv];
        if (mode == 1) {
          float sv = val / (1.f + __expf(-val));
          Cb[(size_t)gm * ldc + col] = (bf16_t)sv;
        } else if (mode == 2) {
          int tok = c_idx[gm];
          Cf[(size_t)tok * ldc + col] += c_scale[gm] * val;
        } else if (mode == 3) {
          Cf[(size_t)gm * ldc + col] = add_src[(size_t)gm * ldc + col] + val;
        } else {
          Cb[(size_t)gm * ldc + col] = (bf16_t)val;
        }
      }
}

// ---------------- Flash attention (causal, online softmax, WMMA) ------------
__global__ __launch_bounds__(128) void attn_kernel(const bf16_t* __restrict__ q,
                                                   const bf16_t* __restrict__ k,
                                                   const bf16_t* __restrict__ v,
                                                   bf16_t* __restrict__ o) {
  int qt = blockIdx.x;           // 64-query tile
  int h  = blockIdx.y;
  int b  = blockIdx.z;
  int qb0 = qt * 64;
  int tid = threadIdx.x;
  int wv_ = tid >> 5;            // 4 waves: 16 queries each
  int lane = tid & 31;
  int g = lane >> 4, ln = lane & 15;

  __shared__ bf16_t Qs[64 * 64];       // [qrow][dim]
  __shared__ bf16_t Ks[32 * 64];       // [key][dim]   (B-frag friendly for QK^T)
  __shared__ bf16_t Vt[64 * 32];       // [dim][key]   (B-frag friendly for PV)
  __shared__ bf16_t Ps[4 * 16 * 32];   // wave-private P tiles

  const size_t hoff = (size_t)h * DHh;
  // stage Q once
#pragma unroll
  for (int c = 0; c < 4; ++c) {
    int idx = tid + 128 * c;             // 0..511
    int r = idx >> 3, c8 = idx & 7;
    v8bf qv = *(const v8bf*)(q + (size_t)(b * Ss + qb0 + r) * Dd + hoff + c8 * 8);
    *(v8bf*)(Qs + r * 64 + c8 * 8) = qv;
  }

  // K/V staging geometry (loop-invariant)
  int sR[2], sC8[2];
  size_t sOff[2];
#pragma unroll
  for (int c = 0; c < 2; ++c) {
    int idx = tid + 128 * c;             // 0..255
    sR[c] = idx >> 3;
    sC8[c] = idx & 7;
    sOff[c] = (size_t)(b * Ss + sR[c]) * Dd + hoff + sC8[c] * 8;
  }
  v8bf kReg[2], vReg[2];
#pragma unroll
  for (int c = 0; c < 2; ++c) {          // prologue: key tile 0
    kReg[c] = *(const v8bf*)(k + sOff[c]);
    vReg[c] = *(const v8bf*)(v + sOff[c]);
  }
  __syncthreads();

  float m_run[8], l_run[8];
  v8f Oacc[4];
#pragma unroll
  for (int i = 0; i < 8; ++i) { m_run[i] = -1e30f; l_run[i] = 0.f; }
#pragma unroll
  for (int nt = 0; nt < 4; ++nt) Oacc[nt] = zero_v8f();

  bf16_t* Pw = Ps + wv_ * 16 * 32;
  int nkt = (qb0 + 64) / 32;
  for (int kt = 0; kt < nkt; ++kt) {
    int ks0 = kt * 32;
    // commit staged K (direct) and V (transposed) registers to LDS
#pragma unroll
    for (int c = 0; c < 2; ++c) {
      *(v8bf*)(Ks + sR[c] * 64 + sC8[c] * 8) = kReg[c];
#pragma unroll
      for (int i = 0; i < 8; ++i) Vt[(sC8[c] * 8 + i) * 32 + sR[c]] = vReg[c][i];
    }
    __syncthreads();

    // issue next tile's global loads; overlap with score/softmax/PV math
    if (kt + 1 < nkt) {
      size_t adv = (size_t)(ks0 + 32) * Dd;
#pragma unroll
      for (int c = 0; c < 2; ++c) {
        kReg[c] = *(const v8bf*)(k + sOff[c] + adv);
        vReg[c] = *(const v8bf*)(v + sOff[c] + adv);
      }
    }

    // scores: S = Q(16x64) K^T(64x32) via 2x2 WMMAs
    v8f s0 = zero_v8f(), s1 = zero_v8f();
    const bf16_t* qrow = Qs + (wv_ * 16 + ln) * 64;
#pragma unroll
    for (int kk = 0; kk < 2; ++kk) {
      v16bf qa = mk_frag(qrow + kk * 32 + 16 * g, qrow + kk * 32 + 16 * g + 8);
      const bf16_t* kr0 = Ks + ln * 64 + kk * 32 + 16 * g;
      const bf16_t* kr1 = Ks + (16 + ln) * 64 + kk * 32 + 16 * g;
      s0 = wmma_bf16(qa, mk_frag(kr0, kr0 + 8), s0);
      s1 = wmma_bf16(qa, mk_frag(kr1, kr1 + 8), s1);
    }

    int qrow0 = qb0 + 16 * wv_ + 8 * g;
#pragma unroll
    for (int vv = 0; vv < 8; ++vv) {
      int qg = qrow0 + vv;
      float a0 = s0[vv] * 0.125f;
      float a1 = s1[vv] * 0.125f;
      if (ks0 + ln > qg)      a0 = -1e30f;
      if (ks0 + 16 + ln > qg) a1 = -1e30f;
      float mx = fmaxf(a0, a1);
#pragma unroll
      for (int off = 8; off >= 1; off >>= 1) mx = fmaxf(mx, __shfl_xor(mx, off, 16));
      float mnew = fmaxf(m_run[vv], mx);
      float al = __expf(m_run[vv] - mnew);
      m_run[vv] = mnew;
      float p0 = __expf(a0 - mnew);
      float p1 = __expf(a1 - mnew);
      float rs = p0 + p1;
#pragma unroll
      for (int off = 8; off >= 1; off >>= 1) rs += __shfl_xor(rs, off, 16);
      l_run[vv] = l_run[vv] * al + rs;
#pragma unroll
      for (int nt = 0; nt < 4; ++nt) Oacc[nt][vv] *= al;
      Pw[(vv + 8 * g) * 32 + ln]      = (bf16_t)p0;   // C-layout -> A-layout via LDS
      Pw[(vv + 8 * g) * 32 + 16 + ln] = (bf16_t)p1;
    }

    // O += P(16x32) V(32x64), 4 WMMAs
#pragma unroll
    for (int nt = 0; nt < 4; ++nt) {
      v16bf pa = mk_frag(Pw + ln * 32 + 16 * g, Pw + ln * 32 + 16 * g + 8);
      const bf16_t* vr = Vt + (nt * 16 + ln) * 32 + 16 * g;
      Oacc[nt] = wmma_bf16(pa, mk_frag(vr, vr + 8), Oacc[nt]);
    }
    __syncthreads();
  }

  // normalize + write
#pragma unroll
  for (int nt = 0; nt < 4; ++nt)
#pragma unroll
    for (int vv = 0; vv < 8; ++vv) {
      int qg = qb0 + 16 * wv_ + 8 * g + vv;
      float val = Oacc[nt][vv] / l_run[vv];
      o[(size_t)(b * Ss + qg) * Dd + hoff + nt * 16 + ln] = (bf16_t)val;
    }
}

// ---------------- router: logits, softmax, top-2, token lists ----------------
__global__ __launch_bounds__(256) void router_kernel(const bf16_t* __restrict__ hn,
                                                     const float* __restrict__ gate,
                                                     float* __restrict__ sumP,
                                                     int* __restrict__ cnt,
                                                     int* __restrict__ tlist,
                                                     float* __restrict__ wl) {
  int tid = threadIdx.x;
  int wv_ = tid >> 5, lane = tid & 31;
  int token = blockIdx.x * 8 + wv_;
  const bf16_t* hr = hn + (size_t)token * Dd;
  float acc[Ee];
#pragma unroll
  for (int e = 0; e < Ee; ++e) acc[e] = 0.f;
  for (int d = lane; d < Dd; d += 32) {
    float xv = (float)hr[d];
    const float* gr = gate + (size_t)d * Ee;
#pragma unroll
    for (int e = 0; e < Ee; ++e) acc[e] += xv * gr[e];
  }
#pragma unroll
  for (int off = 16; off >= 1; off >>= 1)
#pragma unroll
    for (int e = 0; e < Ee; ++e) acc[e] += __shfl_xor(acc[e], off, 32);
  if (lane == 0) {
    float mx = acc[0];
#pragma unroll
    for (int e = 1; e < Ee; ++e) mx = fmaxf(mx, acc[e]);
    float p[Ee], se = 0.f;
#pragma unroll
    for (int e = 0; e < Ee; ++e) { p[e] = __expf(acc[e] - mx); se += p[e]; }
    float inv = 1.f / se;
#pragma unroll
    for (int e = 0; e < Ee; ++e) { p[e] *= inv; atomicAdd(&sumP[e], p[e]); }
    int i0 = 0;
#pragma unroll
    for (int e = 1; e < Ee; ++e) if (p[e] > p[i0]) i0 = e;
    int i1 = (i0 == 0) ? 1 : 0;
#pragma unroll
    for (int e = 0; e < Ee; ++e) if (e != i0 && p[e] > p[i1]) i1 = e;
    float wsum = p[i0] + p[i1];
    int pos0 = atomicAdd(&cnt[i0], 1);
    tlist[i0 * Tt + pos0] = token;
    wl[i0 * Tt + pos0] = p[i0] / wsum;
    int pos1 = atomicAdd(&cnt[i1], 1);
    tlist[i1 * Tt + pos1] = token;
    wl[i1 * Tt + pos1] = p[i1] / wsum;
  }
}

__global__ void zero_kernel(int* cnt, float* sumP) {
  int i = threadIdx.x;
  if (i < Ee) { cnt[i] = 0; sumP[i] = 0.f; }
}

__global__ void loss_kernel(const int* cnt, const float* sumP, float* out_loss) {
  if (threadIdx.x == 0 && blockIdx.x == 0) {
    float acc = 0.f;
    for (int e = 0; e < Ee; ++e) {
      float f = (float)cnt[e] / (float)(Tt * KTOP);
      float P = sumP[e] / (float)Tt;
      acc += f * P;
    }
    *out_loss = (float)Ee * acc;
  }
}

// ---------------- host launch ----------------
extern "C" void kernel_launch(void* const* d_in, const int* in_sizes, int n_in,
                              void* d_out, int out_size, void* d_ws, size_t ws_size,
                              hipStream_t stream) {
  const float* x   = (const float*)d_in[0];
  const float* fc  = (const float*)d_in[1];
  const float* fs  = (const float*)d_in[2];
  const float* anw = (const float*)d_in[3];
  const float* fnw = (const float*)d_in[4];
  const float* wq  = (const float*)d_in[5];
  const float* wk  = (const float*)d_in[6];
  const float* wv  = (const float*)d_in[7];
  const float* wo  = (const float*)d_in[8];
  const float* gw  = (const float*)d_in[9];
  const float* w1  = (const float*)d_in[10];
  const float* w2  = (const float*)d_in[11];
  float* out = (float*)d_out;

  // workspace carve-up (256B aligned)
  size_t off = 0;
  char* base = (char*)d_ws;
  auto carve = [&](size_t bytes) {
    void* p = base + off;
    off += (bytes + 255) & ~(size_t)255;
    return p;
  };
  bf16_t* xnb   = (bf16_t*)carve((size_t)Tt * Dd * 2);
  bf16_t* qb    = (bf16_t*)carve((size_t)Tt * Dd * 2);
  bf16_t* kb    = (bf16_t*)carve((size_t)Tt * Dd * 2);
  bf16_t* vb    = (bf16_t*)carve((size_t)Tt * Dd * 2);
  bf16_t* attnb = (bf16_t*)carve((size_t)Tt * Dd * 2);
  bf16_t* hnb   = (bf16_t*)carve((size_t)Tt * Dd * 2);
  bf16_t* h1b   = (bf16_t*)carve((size_t)Tt * HIDh * 2);
  int*    cnt   = (int*)carve(Ee * sizeof(int));
  float*  sumP  = (float*)carve(Ee * sizeof(float));
  int*    tlist = (int*)carve((size_t)Ee * Tt * sizeof(int));
  float*  wl    = (float*)carve((size_t)Ee * Tt * sizeof(float));

  // 1) attention RMSNorm
  rmsnorm_kernel<<<Tt, 256, 0, stream>>>(x, anw, xnb, 1e-5f);

  // 2) QKV projections (WMMA), bf16 out
  dim3 gproj(Dd / BN, Tt / BM);
  gemm_bf16<<<gproj, 256, 0, stream>>>(xnb, Dd, wq, Dd, nullptr, qb, Dd, Tt, nullptr,
                                       Dd, Dd, nullptr, nullptr, nullptr, nullptr, 4);
  gemm_bf16<<<gproj, 256, 0, stream>>>(xnb, Dd, wk, Dd, nullptr, kb, Dd, Tt, nullptr,
                                       Dd, Dd, nullptr, nullptr, nullptr, nullptr, 4);
  gemm_bf16<<<gproj, 256, 0, stream>>>(xnb, Dd, wv, Dd, nullptr, vb, Dd, Tt, nullptr,
                                       Dd, Dd, nullptr, nullptr, nullptr, nullptr, 4);

  // 3) RoPE in-place on Q, K
  rope_kernel<<<(Tt * Dd / 2) / 256, 256, 0, stream>>>(qb, fc, fs);
  rope_kernel<<<(Tt * Dd / 2) / 256, 256, 0, stream>>>(kb, fc, fs);

  // 4) causal flash attention
  attn_kernel<<<dim3(Ss / 64, NHh, Bb), 128, 0, stream>>>(qb, kb, vb, attnb);

  // 5) output projection + residual -> h (into d_out)
  gemm_bf16<<<gproj, 256, 0, stream>>>(attnb, Dd, wo, Dd, out, nullptr, Dd, Tt,
                                       nullptr, Dd, Dd, nullptr, nullptr, nullptr, x, 3);

  // 6) FFN RMSNorm
  rmsnorm_kernel<<<Tt, 256, 0, stream>>>(out, fnw, hnb, 1e-6f);

  // 7) routing
  zero_kernel<<<1, 32, 0, stream>>>(cnt, sumP);
  router_kernel<<<Tt / 8, 256, 0, stream>>>(hnb, gw, sumP, cnt, tlist, wl);

  // 8) expert FFNs over gathered tokens (top-2 only; blocks early-exit on count)
  for (int e = 0; e < Ee; ++e) {
    gemm_bf16<<<dim3(HIDh / BN, Tt / BM), 256, 0, stream>>>(
        hnb, Dd, w1 + (size_t)e * Dd * HIDh, HIDh, nullptr, h1b, HIDh, 0, cnt + e,
        HIDh, Dd, tlist + e * Tt, nullptr, nullptr, nullptr, 1);
    gemm_bf16<<<dim3(Dd / BN, Tt / BM), 256, 0, stream>>>(
        h1b, HIDh, w2 + (size_t)e * HIDh * Dd, Dd, out, nullptr, Dd, 0, cnt + e,
        Dd, HIDh, nullptr, tlist + e * Tt, wl + e * Tt, nullptr, 2);
  }

  // 9) router loss -> last output element
  loss_kernel<<<1, 32, 0, stream>>>(cnt, sumP, out + (size_t)Tt * Dd);
}
